// GroupedExperts_84396107366902
// MI455X (gfx1250) — matrix operand order repaired and checked by reading, not compiled, not run on hardware
//
#include <hip/hip_runtime.h>
#include <hip/hip_bf16.h>
#include <math.h>

// ---- types for WMMA -------------------------------------------------------
typedef __bf16 bf16_t;
typedef bf16_t v16bf __attribute__((ext_vector_type(16)));
typedef bf16_t v8bf  __attribute__((ext_vector_type(8)));
typedef bf16_t v4bf  __attribute__((ext_vector_type(4)));
typedef float  v8f   __attribute__((ext_vector_type(8)));

#define HIDDEN 1024
#define INTER  768
#define NEXP   64
#define BM 64
#define BN 64
#define BK 64
#define LDA 72          // padded LDS row stride (bf16 elements) = 144B
#define MAX_TILES 192   // sum ceil(c_e/64) <= 8192/64 + 64 = 192

// ---------------------------------------------------------------------------
// Tile table: tbl[0] = count; entry t: {expert, row_start, nrows}
// ---------------------------------------------------------------------------
__global__ __launch_bounds__(32) void build_tile_table(const int* __restrict__ counts,
                                                       int* __restrict__ tbl) {
    if (threadIdx.x != 0) return;
    int cnt = 0, off = 0;
    for (int e = 0; e < NEXP; ++e) {
        int c = counts[e];
        for (int s = 0; s < c; s += BM) {
            tbl[1 + 3 * cnt + 0] = e;
            tbl[1 + 3 * cnt + 1] = off + s;
            tbl[1 + 3 * cnt + 2] = (c - s) < BM ? (c - s) : BM;
            ++cnt;
        }
        off += c;
    }
    tbl[0] = cnt;
}

// Assemble a 16x32 bf16 WMMA operand fragment from two 16-byte LDS chunks.
static __device__ __forceinline__ v16bf frag16(const bf16_t* lo, const bf16_t* hi) {
    v8bf a = *(const v8bf*)lo;
    v8bf b = *(const v8bf*)hi;
    return __builtin_shufflevector(a, b, 0, 1, 2, 3, 4, 5, 6, 7,
                                   8, 9, 10, 11, 12, 13, 14, 15);
}

// ---------------------------------------------------------------------------
// Phase 1: h = silu(x @ w1) * (x @ w3), per expert tile, h stored bf16 in ws
// grid = (INTER/BN, MAX_TILES), block = 256 (8 waves)
// ---------------------------------------------------------------------------
__global__ __launch_bounds__(256) void moe_gateup_kernel(
        const float* __restrict__ x, const float* __restrict__ w1,
        const float* __restrict__ w3, const int* __restrict__ tbl,
        bf16_t* __restrict__ hbuf) {
    __shared__ bf16_t lds_a [2][BM * LDA];
    __shared__ bf16_t lds_b1[2][BN * LDA];
    __shared__ bf16_t lds_b3[2][BN * LDA];

    const int tile = blockIdx.y;
    if (tile >= tbl[0]) return;
    const int e     = tbl[1 + 3 * tile + 0];
    const int row0  = tbl[1 + 3 * tile + 1];
    const int nrows = tbl[1 + 3 * tile + 2];
    const int nb    = blockIdx.x * BN;

    const int tid   = threadIdx.x;
    const int lane  = tid & 31;
    const int wave  = tid >> 5;
    const int rl    = lane & 15;
    const int hi    = lane >> 4;
    const int msub  = wave & 3;      // M subtile 0..3
    const int npair = wave >> 2;     // N pair 0..1

    // A staging role: one row, 16 consecutive k
    const int ar   = tid >> 2;            // 0..63
    const int aks  = (tid & 3) * 16;      // 0,16,32,48
    const bool rok = ar < nrows;
    const float* aptr = x + (size_t)(row0 + ar) * HIDDEN + aks;

    // B staging role: 4(k) x 4(n) block -> packed ds_store_b64 (transposed)
    const int k0 = (tid >> 4) * 4;        // 0..60
    const int n0 = (tid & 15) * 4;        // 0..60
    const size_t wbase = (size_t)e * HIDDEN * INTER;
    const float* b1p = w1 + wbase + (size_t)k0 * INTER + nb + n0;
    const float* b3p = w3 + wbase + (size_t)k0 * INTER + nb + n0;

    float fa[16], f1[16], f3[16];

    auto gload = [&](int kb) {
        if (rok) {
            const float4* p = (const float4*)(aptr + kb);
            #pragma unroll
            for (int q = 0; q < 4; ++q) {
                float4 v = p[q];
                fa[4 * q + 0] = v.x; fa[4 * q + 1] = v.y;
                fa[4 * q + 2] = v.z; fa[4 * q + 3] = v.w;
            }
        } else {
            #pragma unroll
            for (int q = 0; q < 16; ++q) fa[q] = 0.0f;
        }
        #pragma unroll
        for (int i = 0; i < 4; ++i) {
            float4 a = *(const float4*)(b1p + (size_t)(kb + i) * INTER);
            float4 b = *(const float4*)(b3p + (size_t)(kb + i) * INTER);
            f1[4 * i + 0] = a.x; f1[4 * i + 1] = a.y;
            f1[4 * i + 2] = a.z; f1[4 * i + 3] = a.w;
            f3[4 * i + 0] = b.x; f3[4 * i + 1] = b.y;
            f3[4 * i + 2] = b.z; f3[4 * i + 3] = b.w;
        }
    };

    auto stage = [&](int buf) {
        v8bf p0 = {}, p1 = {};
        #pragma unroll
        for (int q = 0; q < 8; ++q) {
            p0[q] = (bf16_t)fa[q];
            p1[q] = (bf16_t)fa[q + 8];
        }
        *(v8bf*)&lds_a[buf][ar * LDA + aks]     = p0;
        *(v8bf*)&lds_a[buf][ar * LDA + aks + 8] = p1;
        #pragma unroll
        for (int n = 0; n < 4; ++n) {
            v4bf q1 = {}, q3 = {};
            #pragma unroll
            for (int i = 0; i < 4; ++i) {
                q1[i] = (bf16_t)f1[4 * i + n];   // consecutive k -> packed
                q3[i] = (bf16_t)f3[4 * i + n];
            }
            *(v4bf*)&lds_b1[buf][(n0 + n) * LDA + k0] = q1;
            *(v4bf*)&lds_b3[buf][(n0 + n) * LDA + k0] = q3;
        }
    };

    v8f cg[2] = {};
    v8f cu[2] = {};

    gload(0);
    int buf = 0;
    for (int kb = 0; kb < HIDDEN; kb += BK) {
        stage(buf);
        if (kb + BK < HIDDEN) {
            gload(kb + BK);                       // overlaps barrier + WMMA burst
            if (kb + 2 * BK < HIDDEN) {
                __builtin_prefetch(b1p + (size_t)(kb + 2 * BK) * INTER, 0, 0);
                __builtin_prefetch(b3p + (size_t)(kb + 2 * BK) * INTER, 0, 0);
            }
        }
        __syncthreads();
        #pragma unroll
        for (int ks = 0; ks < BK; ks += 32) {
            const int arow = msub * 16 + rl;
            v16bf af = frag16(&lds_a[buf][arow * LDA + ks + hi * 8],
                              &lds_a[buf][arow * LDA + ks + 16 + hi * 8]);
            #pragma unroll
            for (int j = 0; j < 2; ++j) {
                const int bcol = (npair * 2 + j) * 16 + rl;
                v16bf fb1 = frag16(&lds_b1[buf][bcol * LDA + ks + hi * 8],
                                   &lds_b1[buf][bcol * LDA + ks + 16 + hi * 8]);
                v16bf fb3 = frag16(&lds_b3[buf][bcol * LDA + ks + hi * 8],
                                   &lds_b3[buf][bcol * LDA + ks + 16 + hi * 8]);
                cg[j] = __builtin_amdgcn_wmma_f32_16x16x32_bf16(
                            false, af, false, fb1, (short)0, cg[j], false, false);
                cu[j] = __builtin_amdgcn_wmma_f32_16x16x32_bf16(
                            false, af, false, fb3, (short)0, cu[j], false, false);
            }
        }
        buf ^= 1;
    }

    // ---- epilogue: h = silu(gate) * up, bf16 store to workspace
    #pragma unroll
    for (int j = 0; j < 2; ++j) {
        const int ncol = nb + (npair * 2 + j) * 16 + rl;
        #pragma unroll
        for (int i = 0; i < 8; ++i) {
            const int rloc = msub * 16 + i + hi * 8;
            if (rloc < nrows) {
                float g = cg[j][i];
                float u = cu[j][i];
                float s = g / (1.0f + __expf(-g));
                hbuf[(size_t)(row0 + rloc) * INTER + ncol] = (bf16_t)(s * u);
            }
        }
    }
}

// ---------------------------------------------------------------------------
// Phase 2: out = h @ w2 per expert tile, fp32 output.
// A operand (bf16 h, L2-resident) staged with GLOBAL_LOAD_ASYNC_TO_LDS_B128.
// grid = (HIDDEN/BN, MAX_TILES), block = 256
// ---------------------------------------------------------------------------
__global__ __launch_bounds__(256) void moe_down_kernel(
        const bf16_t* __restrict__ hbuf, const float* __restrict__ w2,
        float* __restrict__ out, const int* __restrict__ tbl) {
    __shared__ bf16_t lds_a[2][BM * LDA];
    __shared__ bf16_t lds_b[2][BN * LDA];

    const int tile = blockIdx.y;
    if (tile >= tbl[0]) return;
    const int e     = tbl[1 + 3 * tile + 0];
    const int row0  = tbl[1 + 3 * tile + 1];
    const int nrows = tbl[1 + 3 * tile + 2];
    const int nb    = blockIdx.x * BN;

    const int tid   = threadIdx.x;
    const int lane  = tid & 31;
    const int wave  = tid >> 5;
    const int rl    = lane & 15;
    const int hi    = lane >> 4;
    const int msub  = wave & 3;
    const int npair = wave >> 2;

    const int ar   = tid >> 2;
    const int aks  = (tid & 3) * 16;
    const bool rok = ar < nrows;
    const bf16_t* aptr = hbuf + (size_t)(row0 + ar) * INTER + aks;

    const int k0 = (tid >> 4) * 4;
    const int n0 = (tid & 15) * 4;
    const size_t wbase = (size_t)e * INTER * HIDDEN;
    const float* bp = w2 + wbase + (size_t)k0 * HIDDEN + nb + n0;

    float fb[16];

    auto gloadB = [&](int kb) {
        #pragma unroll
        for (int i = 0; i < 4; ++i) {
            float4 v = *(const float4*)(bp + (size_t)(kb + i) * HIDDEN);
            fb[4 * i + 0] = v.x; fb[4 * i + 1] = v.y;
            fb[4 * i + 2] = v.z; fb[4 * i + 3] = v.w;
        }
    };

    // Async copy of 32 bytes of bf16 h straight into LDS (no VGPR round-trip).
    auto asyncA = [&](int buf, int kb) {
        if (rok) {
            unsigned l0 = (unsigned)(size_t)&lds_a[buf][ar * LDA + aks];
            unsigned long long g0 = (unsigned long long)(size_t)(aptr + kb);
            asm volatile(
                "global_load_async_to_lds_b128 %0, %1, off\n\t"
                "global_load_async_to_lds_b128 %2, %3, off"
                :: "v"(l0), "v"(g0), "v"(l0 + 16u), "v"(g0 + 16ull)
                : "memory");
        }
    };

    auto stageB = [&](int buf) {
        #pragma unroll
        for (int n = 0; n < 4; ++n) {
            v4bf q = {};
            #pragma unroll
            for (int i = 0; i < 4; ++i) q[i] = (bf16_t)fb[4 * i + n];
            *(v4bf*)&lds_b[buf][(n0 + n) * LDA + k0] = q;
        }
    };

    v8f c[2] = {};

    gloadB(0);
    int buf = 0;
    for (int kb = 0; kb < INTER; kb += BK) {
        asyncA(buf, kb);                 // ASYNCcnt-tracked global -> LDS
        stageB(buf);
        if (kb + BK < INTER) {
            gloadB(kb + BK);
            if (kb + 2 * BK < INTER) {
                __builtin_prefetch(bp + (size_t)(kb + 2 * BK) * HIDDEN, 0, 0);
            }
        }
        asm volatile("s_wait_asynccnt 0x0" ::: "memory");
        __syncthreads();
        #pragma unroll
        for (int ks = 0; ks < BK; ks += 32) {
            const int arow = msub * 16 + rl;
            v16bf af = frag16(&lds_a[buf][arow * LDA + ks + hi * 8],
                              &lds_a[buf][arow * LDA + ks + 16 + hi * 8]);
            #pragma unroll
            for (int j = 0; j < 2; ++j) {
                const int bcol = (npair * 2 + j) * 16 + rl;
                v16bf fbv = frag16(&lds_b[buf][bcol * LDA + ks + hi * 8],
                                   &lds_b[buf][bcol * LDA + ks + 16 + hi * 8]);
                c[j] = __builtin_amdgcn_wmma_f32_16x16x32_bf16(
                           false, af, false, fbv, (short)0, c[j], false, false);
            }
        }
        buf ^= 1;
    }

    // ---- epilogue: fp32 store
    #pragma unroll
    for (int j = 0; j < 2; ++j) {
        const int ncol = nb + (npair * 2 + j) * 16 + rl;
        #pragma unroll
        for (int i = 0; i < 8; ++i) {
            const int rloc = msub * 16 + i + hi * 8;
            if (rloc < nrows) {
                out[(size_t)(row0 + rloc) * HIDDEN + ncol] = c[j][i];
            }
        }
    }
}

// ---------------------------------------------------------------------------
extern "C" void kernel_launch(void* const* d_in, const int* in_sizes, int n_in,
                              void* d_out, int out_size, void* d_ws, size_t ws_size,
                              hipStream_t stream) {
    (void)in_sizes; (void)n_in; (void)out_size; (void)ws_size;
    const float* x      = (const float*)d_in[0];
    const float* w1     = (const float*)d_in[1];
    const float* w2     = (const float*)d_in[2];
    const float* w3     = (const float*)d_in[3];
    const int*   counts = (const int*)d_in[4];
    float*       out    = (float*)d_out;

    // workspace layout: [0,4096) tile table; [4096, ...) bf16 h buffer (12.6 MB)
    int*    tbl  = (int*)d_ws;
    bf16_t* hbuf = (bf16_t*)((char*)d_ws + 4096);

    build_tile_table<<<1, 32, 0, stream>>>(counts, tbl);
    moe_gateup_kernel<<<dim3(INTER / BN, MAX_TILES), 256, 0, stream>>>(
        x, w1, w3, tbl, hbuf);
    moe_down_kernel<<<dim3(HIDDEN / BN, MAX_TILES), 256, 0, stream>>>(
        hbuf, w2, out, tbl);
}
